// MPMC_net_83708912599362
// MI455X (gfx1250) — matrix operand order repaired and checked by reading, not compiled, not run on hardware
//
#include <hip/hip_runtime.h>
#include <hip/hip_bf16.h>

#define NB      16
#define NS      2048
#define NN      (NB * NS)        // 32768 nodes
#define NHID    64
#define NLAYERS 4
#define EPG     32768
#define NE      (NB * EPG)       // 524288 edges
#define EPSN    1e-5f
#define PI_D    3.14159265358979323846

typedef __attribute__((ext_vector_type(16))) __bf16 v16bf;
typedef __attribute__((ext_vector_type(8)))  float  v8f;

union Frag16 {
    v16bf  v;
    uint4  q[2];
    __bf16 h[16];
};

__device__ __forceinline__ v8f wmma_bf16(const Frag16& a, const Frag16& b, v8f c) {
    return __builtin_amdgcn_wmma_f32_16x16x32_bf16(false, a.v, false, b.v,
                                                   (short)0, c, false, false);
}

// B fragment: 16 contiguous bf16 (packed by pack_kernel into WMMA B lane layout)
__device__ __forceinline__ Frag16 load_bfrag(const __bf16* wp) {
    Frag16 b;
    b.q[0] = *reinterpret_cast<const uint4*>(wp);
    b.q[1] = *reinterpret_cast<const uint4*>(wp + 8);
    return b;
}

// A fragment from a bf16 row: elems 0..7 at rowp, elems 8..15 at rowp+16 (K gap 16)
__device__ __forceinline__ Frag16 load_afrag_bf16(const __bf16* rowp) {
    Frag16 a;
    a.q[0] = *reinterpret_cast<const uint4*>(rowp);
    a.q[1] = *reinterpret_cast<const uint4*>(rowp + 16);
    return a;
}

// A fragment from an f32 row (convert to bf16)
__device__ __forceinline__ Frag16 load_afrag_f32(const float* rowp) {
    Frag16 a;
#pragma unroll
    for (int e = 0; e < 8; ++e) {
        a.h[e]     = (__bf16)rowp[e];
        a.h[8 + e] = (__bf16)rowp[16 + e];
    }
    return a;
}

__device__ __forceinline__ void atomAddF(float* p, float v) { unsafeAtomicAdd(p, v); }

// -------------------------------------------------------------------------
// Pack f32 weights [L][K][64] -> bf16 WMMA B-operand layout:
// tile(c=k/32, nt=n/16): lane = (((k%32)>>3)&1)*16 + n%16, elem = ((k%32)&7) + ((k%32)>>4)*8
// -------------------------------------------------------------------------
__global__ void pack_kernel(const float* __restrict__ src, __bf16* __restrict__ dst,
                            int K, int total) {
    int tid = blockIdx.x * blockDim.x + threadIdx.x;
    if (tid >= total) return;
    int layer = tid / (K * 64);
    int rem   = tid - layer * K * 64;
    int k     = rem / 64;
    int n     = rem - k * 64;
    int c     = k >> 5;
    int kin   = k & 31;
    int ntile = n >> 4;
    int ncol  = n & 15;
    int lane  = ((kin >> 3) & 1) * 16 + ncol;
    int e     = (kin & 7) + ((kin >> 4) & 1) * 8;
    dst[layer * K * 64 + ((c * 4 + ntile) * 32 + lane) * 16 + e] = (__bf16)src[tid];
}

__global__ void zero_kernel(float* __restrict__ p, int n) {
    int tid = blockIdx.x * blockDim.x + threadIdx.x;
    if (tid < n) p[tid] = 0.0f;
}

// h = x @ enc_W + enc_b  -> bf16 [N][64]
__global__ void enc_kernel(const float* __restrict__ x, const float* __restrict__ W,
                           const float* __restrict__ b, __bf16* __restrict__ h) {
    int tid = blockIdx.x * blockDim.x + threadIdx.x;
    if (tid >= NN * NHID) return;
    int n = tid >> 6, c = tid & 63;
    float v = b[c] + x[n * 3 + 0] * W[c] + x[n * 3 + 1] * W[64 + c] + x[n * 3 + 2] * W[128 + c];
    h[tid] = (__bf16)v;
}

// -------------------------------------------------------------------------
// Edge MLP + scatter-add:  m = relu(relu([h[dst]|h[src]]@W1+b1)@W2+b2);
// agg[dst] += m.  One wave = 16 edges. 24 WMMAs / tile.
// -------------------------------------------------------------------------
__global__ __launch_bounds__(256)
void edge_kernel(const __bf16* __restrict__ h,
                 const int* __restrict__ esrc, const int* __restrict__ edst,
                 const __bf16* __restrict__ w1p, const float* __restrict__ b1,
                 const __bf16* __restrict__ w2p, const float* __restrict__ b2,
                 float* __restrict__ agg, int ntiles) {
    __shared__ __bf16 smem[8 * 16 * 64];
    const int lane  = threadIdx.x & 31;
    const int wave  = threadIdx.x >> 5;
    __bf16*   lm    = smem + wave * (16 * 64);
    const int col   = lane & 15;
    const int hi    = (lane >> 4) & 1;
    const int rbase = hi * 8;
    const int wg    = blockIdx.x * (blockDim.x >> 5) + wave;
    const int nw    = gridDim.x * (blockDim.x >> 5);

    for (int t = wg; t < ntiles; t += nw) {
        const int e0 = t * 16;
        const int nd = edst[e0 + col];   // xi = h[edge_dst]
        const int ns = esrc[e0 + col];   // xj = h[edge_src]
        int dstIdx[8];
#pragma unroll
        for (int r = 0; r < 8; ++r) dstIdx[r] = edst[e0 + rbase + r];

        // GEMM1: [16,128]@[128,64], bias preloaded
        v8f acc[4];
#pragma unroll
        for (int n = 0; n < 4; ++n) {
            float bv = b1[n * 16 + col];
#pragma unroll
            for (int r = 0; r < 8; ++r) acc[n][r] = bv;
        }
#pragma unroll
        for (int c = 0; c < 4; ++c) {
            const int node = (c < 2) ? nd : ns;
            const int kr   = (c & 1) * 32 + hi * 8;
            Frag16 a = load_afrag_bf16(h + (long)node * 64 + kr);
#pragma unroll
            for (int n = 0; n < 4; ++n) {
                Frag16 b = load_bfrag(w1p + ((c * 4 + n) * 32 + lane) * 16);
                acc[n]   = wmma_bf16(a, b, acc[n]);
            }
        }
        // ReLU + transpose to LDS (D layout: lane=col, rows in VGPRs)
#pragma unroll
        for (int n = 0; n < 4; ++n)
#pragma unroll
            for (int r = 0; r < 8; ++r)
                lm[(rbase + r) * 64 + n * 16 + col] = (__bf16)fmaxf(acc[n][r], 0.0f);

        // GEMM2: [16,64]@[64,64]
        v8f acc2[4];
#pragma unroll
        for (int n = 0; n < 4; ++n) {
            float bv = b2[n * 16 + col];
#pragma unroll
            for (int r = 0; r < 8; ++r) acc2[n][r] = bv;
        }
#pragma unroll
        for (int c = 0; c < 2; ++c) {
            Frag16 a = load_afrag_bf16(lm + col * 64 + c * 32 + rbase);
#pragma unroll
            for (int n = 0; n < 4; ++n) {
                Frag16 b = load_bfrag(w2p + ((c * 4 + n) * 32 + lane) * 16);
                acc2[n]  = wmma_bf16(a, b, acc2[n]);
            }
        }
        // ReLU + segment_sum via native f32 atomics
#pragma unroll
        for (int n = 0; n < 4; ++n)
#pragma unroll
            for (int r = 0; r < 8; ++r)
                atomAddF(&agg[(long)dstIdx[r] * 64 + n * 16 + col],
                         fmaxf(acc2[n][r], 0.0f));
    }
}

// -------------------------------------------------------------------------
// Node update: u = relu(relu([h|agg]@U1+b1)@U2+b2) -> f32 [N][64]
// -------------------------------------------------------------------------
__global__ __launch_bounds__(256)
void node_kernel(const __bf16* __restrict__ h, const float* __restrict__ agg,
                 const __bf16* __restrict__ w1p, const float* __restrict__ b1,
                 const __bf16* __restrict__ w2p, const float* __restrict__ b2,
                 float* __restrict__ u, int ntiles) {
    __shared__ __bf16 smem[8 * 16 * 64];
    const int lane  = threadIdx.x & 31;
    const int wave  = threadIdx.x >> 5;
    __bf16*   lm    = smem + wave * (16 * 64);
    const int col   = lane & 15;
    const int hi    = (lane >> 4) & 1;
    const int rbase = hi * 8;
    const int wg    = blockIdx.x * (blockDim.x >> 5) + wave;
    const int nw    = gridDim.x * (blockDim.x >> 5);

    for (int t = wg; t < ntiles; t += nw) {
        const int n0   = t * 16;
        const int node = n0 + col;

        v8f acc[4];
#pragma unroll
        for (int n = 0; n < 4; ++n) {
            float bv = b1[n * 16 + col];
#pragma unroll
            for (int r = 0; r < 8; ++r) acc[n][r] = bv;
        }
#pragma unroll
        for (int c = 0; c < 4; ++c) {
            const int kr = (c & 1) * 32 + hi * 8;
            Frag16 a = (c < 2) ? load_afrag_bf16(h + (long)node * 64 + kr)
                               : load_afrag_f32(agg + (long)node * 64 + kr);
#pragma unroll
            for (int n = 0; n < 4; ++n) {
                Frag16 b = load_bfrag(w1p + ((c * 4 + n) * 32 + lane) * 16);
                acc[n]   = wmma_bf16(a, b, acc[n]);
            }
        }
#pragma unroll
        for (int n = 0; n < 4; ++n)
#pragma unroll
            for (int r = 0; r < 8; ++r)
                lm[(rbase + r) * 64 + n * 16 + col] = (__bf16)fmaxf(acc[n][r], 0.0f);

        v8f acc2[4];
#pragma unroll
        for (int n = 0; n < 4; ++n) {
            float bv = b2[n * 16 + col];
#pragma unroll
            for (int r = 0; r < 8; ++r) acc2[n][r] = bv;
        }
#pragma unroll
        for (int c = 0; c < 2; ++c) {
            Frag16 a = load_afrag_bf16(lm + col * 64 + c * 32 + rbase);
#pragma unroll
            for (int n = 0; n < 4; ++n) {
                Frag16 b = load_bfrag(w2p + ((c * 4 + n) * 32 + lane) * 16);
                acc2[n]  = wmma_bf16(a, b, acc2[n]);
            }
        }
#pragma unroll
        for (int n = 0; n < 4; ++n)
#pragma unroll
            for (int r = 0; r < 8; ++r)
                u[(long)(n0 + rbase + r) * 64 + n * 16 + col] = fmaxf(acc2[n][r], 0.0f);
    }
}

// InstanceNorm stats per (graph, channel): biased var over 2048 nodes
__global__ void stats_kernel(const float* __restrict__ u, float* __restrict__ mean,
                             float* __restrict__ rstd) {
    __shared__ float s1[256], s2[256];
    const int b = blockIdx.x, c = blockIdx.y, t = threadIdx.x;
    const float* up = u + (long)(b * NS) * 64 + c;
    float s = 0.f, q = 0.f;
    for (int i = t; i < NS; i += 256) {
        float v = up[(long)i * 64];
        s += v; q += v * v;
    }
    s1[t] = s; s2[t] = q;
    __syncthreads();
    for (int k = 128; k > 0; k >>= 1) {
        if (t < k) { s1[t] += s1[t + k]; s2[t] += s2[t + k]; }
        __syncthreads();
    }
    if (t == 0) {
        float m   = s1[0] / (float)NS;
        float var = s2[0] / (float)NS - m * m;
        mean[b * 64 + c] = m;
        rstd[b * 64 + c] = rsqrtf(var + EPSN);
    }
}

__global__ void norm_kernel(const float* __restrict__ u, const float* __restrict__ mean,
                            const float* __restrict__ rstd, __bf16* __restrict__ h) {
    int tid = blockIdx.x * blockDim.x + threadIdx.x;
    if (tid >= NN * NHID) return;
    int n = tid >> 6, c = tid & 63, b = n >> 11;
    h[tid] = (__bf16)((u[tid] - mean[b * 64 + c]) * rstd[b * 64 + c]);
}

// X = normalize_rows(h @ dec_W + dec_b)
__global__ void dec_kernel(const __bf16* __restrict__ h, const float* __restrict__ W,
                           const float* __restrict__ b, float* __restrict__ X) {
    int n = blockIdx.x * blockDim.x + threadIdx.x;
    if (n >= NN) return;
    const __bf16* hr = h + (long)n * 64;
    float a0 = b[0], a1 = b[1], a2 = b[2];
#pragma unroll
    for (int k = 0; k < 64; ++k) {
        float hv = (float)hr[k];
        a0 += hv * W[k * 3 + 0];
        a1 += hv * W[k * 3 + 1];
        a2 += hv * W[k * 3 + 2];
    }
    float inv = 1.0f / sqrtf(a0 * a0 + a1 * a1 + a2 * a2);
    X[n * 3 + 0] = a0 * inv;
    X[n * 3 + 1] = a1 * inv;
    X[n * 3 + 2] = a2 * inv;
}

__global__ void y_kernel(float* __restrict__ Y) {
    int i = threadIdx.x;  // 64 threads
    double phi = PI_D * (3.0 - sqrt(5.0));
    double y   = 1.0 - 2.0 * (double)i / 63.0;
    double r   = sqrt(fmax(0.0, 1.0 - y * y));
    double th  = phi * (double)i;
    Y[i * 3 + 0] = (float)(cos(th) * r);
    Y[i * 3 + 1] = (float)y;
    Y[i * 3 + 2] = (float)(sin(th) * r);
}

// kXX / kXY partial sums per graph. 128 blocks: blockIdx.x = b*8 + rowchunk
__global__ __launch_bounds__(256)
void loss_kernel(const float* __restrict__ X, const float* __restrict__ Y,
                 float* __restrict__ kxx, float* __restrict__ kxy) {
    __shared__ float sX[256 * 3];
    __shared__ float sY[64 * 3];
    __shared__ float red[256];
    const int b  = blockIdx.x >> 3;
    const int rc = blockIdx.x & 7;
    const int t  = threadIdx.x;
    const float* Xg = X + (long)b * NS * 3;
    if (t < 64) {
        sY[t * 3 + 0] = Y[t * 3 + 0];
        sY[t * 3 + 1] = Y[t * 3 + 1];
        sY[t * 3 + 2] = Y[t * 3 + 2];
    }
    const int i = rc * 256 + t;
    const float xi0 = Xg[i * 3 + 0], xi1 = Xg[i * 3 + 1], xi2 = Xg[i * 3 + 2];
    float accXX = 0.f;
    for (int jt = 0; jt < 8; ++jt) {
        __syncthreads();
        const int j = jt * 256 + t;
        sX[t * 3 + 0] = Xg[j * 3 + 0];
        sX[t * 3 + 1] = Xg[j * 3 + 1];
        sX[t * 3 + 2] = Xg[j * 3 + 2];
        __syncthreads();
        for (int jj = 0; jj < 256; ++jj) {
            float d = xi0 * sX[jj * 3] + xi1 * sX[jj * 3 + 1] + xi2 * sX[jj * 3 + 2];
            accXX += __expf((d - 1.0f) * 2.0f);
        }
    }
    float accXY = 0.f;
#pragma unroll 4
    for (int m = 0; m < 64; ++m) {
        float d = xi0 * sY[m * 3] + xi1 * sY[m * 3 + 1] + xi2 * sY[m * 3 + 2];
        accXY += __expf((d - 1.0f) * 2.0f);
    }
    red[t] = accXX;
    __syncthreads();
    for (int s = 128; s > 0; s >>= 1) {
        if (t < s) red[t] += red[t + s];
        __syncthreads();
    }
    if (t == 0) atomAddF(&kxx[b], red[0]);
    __syncthreads();
    red[t] = accXY;
    __syncthreads();
    for (int s = 128; s > 0; s >>= 1) {
        if (t < s) red[t] += red[t + s];
        __syncthreads();
    }
    if (t == 0) atomAddF(&kxy[b], red[0]);
}

__global__ void finalize_kernel(const float* __restrict__ Y, const float* __restrict__ kxx,
                                const float* __restrict__ kxy, float* __restrict__ out0) {
    __shared__ float red[64];
    int t = threadIdx.x;  // 64 threads
    float y0 = Y[t * 3 + 0], y1 = Y[t * 3 + 1], y2 = Y[t * 3 + 2];
    float acc = 0.f;
    for (int m = 0; m < 64; ++m) {
        float d = y0 * Y[m * 3] + y1 * Y[m * 3 + 1] + y2 * Y[m * 3 + 2];
        acc += __expf((d - 1.0f) * 2.0f);
    }
    red[t] = acc;
    __syncthreads();
    for (int s = 32; s > 0; s >>= 1) {
        if (t < s) red[t] += red[t + s];
        __syncthreads();
    }
    if (t == 0) {
        float kyy  = red[0] / (64.0f * 64.0f);
        float loss = 0.f;
        for (int b = 0; b < NB; ++b)
            loss += kxx[b] / ((float)NS * (float)NS)
                  - 2.0f * kxy[b] / ((float)NS * 64.0f) + kyy;
        out0[0] = loss / (float)NB;
    }
}

// -------------------------------------------------------------------------
extern "C" void kernel_launch(void* const* d_in, const int* in_sizes, int n_in,
                              void* d_out, int out_size, void* d_ws, size_t ws_size,
                              hipStream_t stream) {
    const float* x     = (const float*)d_in[0];
    const int*   esrc  = (const int*)d_in[1];
    const int*   edst  = (const int*)d_in[2];
    const float* enc_W = (const float*)d_in[4];
    const float* enc_b = (const float*)d_in[5];
    const float* m1W   = (const float*)d_in[6];
    const float* m1b   = (const float*)d_in[7];
    const float* m2W   = (const float*)d_in[8];
    const float* m2b   = (const float*)d_in[9];
    const float* u1W   = (const float*)d_in[10];
    const float* u1b   = (const float*)d_in[11];
    const float* u2W   = (const float*)d_in[12];
    const float* u2b   = (const float*)d_in[13];
    const float* decW  = (const float*)d_in[14];
    const float* decb  = (const float*)d_in[15];
    float* out = (float*)d_out;

    char* ws = (char*)d_ws;
    size_t off = 0;
    auto take = [&](size_t bytes) -> char* {
        char* p = ws + off;
        off += (bytes + 255) & ~(size_t)255;
        return p;
    };
    __bf16* hbuf  = (__bf16*)take((size_t)NN * 64 * 2);
    float*  agg   = (float*)take((size_t)NN * 64 * 4);
    float*  ubuf  = (float*)take((size_t)NN * 64 * 4);
    __bf16* m1p   = (__bf16*)take((size_t)4 * 128 * 64 * 2);
    __bf16* m2p   = (__bf16*)take((size_t)4 * 64 * 64 * 2);
    __bf16* u1p   = (__bf16*)take((size_t)4 * 128 * 64 * 2);
    __bf16* u2p   = (__bf16*)take((size_t)4 * 64 * 64 * 2);
    float*  meanp = (float*)take((size_t)NB * 64 * 4);
    float*  rstdp = (float*)take((size_t)NB * 64 * 4);
    float*  Yp    = (float*)take((size_t)64 * 3 * 4);
    float*  accp  = (float*)take((size_t)32 * 4);  // kxx[16] | kxy[16]

    // Pack weights into WMMA B-operand layout (bf16)
    pack_kernel<<<(4 * 128 * 64 + 255) / 256, 256, 0, stream>>>(m1W, m1p, 128, 4 * 128 * 64);
    pack_kernel<<<(4 * 64 * 64 + 255) / 256, 256, 0, stream>>>(m2W, m2p, 64, 4 * 64 * 64);
    pack_kernel<<<(4 * 128 * 64 + 255) / 256, 256, 0, stream>>>(u1W, u1p, 128, 4 * 128 * 64);
    pack_kernel<<<(4 * 64 * 64 + 255) / 256, 256, 0, stream>>>(u2W, u2p, 64, 4 * 64 * 64);

    // Encoder
    enc_kernel<<<(NN * 64 + 255) / 256, 256, 0, stream>>>(x, enc_W, enc_b, hbuf);

    for (int l = 0; l < NLAYERS; ++l) {
        zero_kernel<<<(NN * 64 + 255) / 256, 256, 0, stream>>>(agg, NN * 64);
        edge_kernel<<<2048, 256, 0, stream>>>(hbuf, esrc, edst,
                                              m1p + (size_t)l * 128 * 64, m1b + l * 64,
                                              m2p + (size_t)l * 64 * 64, m2b + l * 64,
                                              agg, NE / 16);
        node_kernel<<<256, 256, 0, stream>>>(hbuf, agg,
                                             u1p + (size_t)l * 128 * 64, u1b + l * 64,
                                             u2p + (size_t)l * 64 * 64, u2b + l * 64,
                                             ubuf, NN / 16);
        stats_kernel<<<dim3(NB, 64), 256, 0, stream>>>(ubuf, meanp, rstdp);
        norm_kernel<<<(NN * 64 + 255) / 256, 256, 0, stream>>>(ubuf, meanp, rstdp, hbuf);
    }

    // Decoder -> X goes straight into d_out[1..]
    dec_kernel<<<(NN + 255) / 256, 256, 0, stream>>>(hbuf, decW, decb, out + 1);

    // Loss
    y_kernel<<<1, 64, 0, stream>>>(Yp);
    zero_kernel<<<1, 32, 0, stream>>>(accp, 32);
    loss_kernel<<<NB * 8, 256, 0, stream>>>(out + 1, Yp, accp, accp + 16);
    finalize_kernel<<<1, 64, 0, stream>>>(Yp, accp, accp + 16, out);
}
